// StandaloneAlignment_4183298146472
// MI455X (gfx1250) — matrix operand order repaired and checked by reading, not compiled, not run on hardware
//
#include <hip/hip_runtime.h>
#include <hip/hip_bf16.h>
#include <cstdint>
#include <cstddef>

// ---------------- problem constants ----------------
#define BB     16
#define T1C    2048
#define T2C    512
#define NMEL   100
#define NTEXT  512
#define CC     128
#define HIDC   384
#define MAXPOS 8192
#define POSA   0.31622776601683794f   // sqrt(0.1)

typedef __attribute__((ext_vector_type(16))) _Float16 v16h;
typedef __attribute__((ext_vector_type(8)))  float    v8f;

// Async global->LDS 16B copy (CDNA5 GLOBAL_LOAD_ASYNC_TO_LDS_B128, ASYNCcnt-tracked).
// ldsaddr: 32-bit LDS byte address (low 32 bits of the flat shared pointer),
// gaddr:   64-bit global byte address.
#define ASYNC_COPY_B128(ldsaddr, gaddr)                                    \
  asm volatile("global_load_async_to_lds_b128 %0, %1, off"                 \
               :: "v"(ldsaddr), "v"(gaddr) : "memory")
#define WAIT_ASYNCCNT0() asm volatile("s_wait_asynccnt 0" ::: "memory")

// ---------------- helpers ----------------
__device__ __forceinline__ float blockReduceSum(float v, float* sred) {
#pragma unroll
  for (int m = 16; m; m >>= 1) v += __shfl_xor(v, m, 32);
  int wid = threadIdx.x >> 5;
  int lane = threadIdx.x & 31;
  __syncthreads();                 // protect sred reuse across calls
  if (lane == 0) sred[wid] = v;
  __syncthreads();
  int nw = (blockDim.x + 31) >> 5;
  float r = (threadIdx.x < (unsigned)nw) ? sred[threadIdx.x] : 0.f;
#pragma unroll
  for (int m = 16; m; m >>= 1) r += __shfl_xor(r, m, 32);
  if (threadIdx.x == 0) sred[0] = r;
  __syncthreads();
  return sred[0];
}

// A-matrix fragment (16x32 f16, MxK): lane half h holds K = {h*8..h*8+7} U {16+h*8..16+h*8+7}
__device__ __forceinline__ v16h load_fragA(const _Float16* tile, int ld, int lane) {
  const _Float16* p = tile + (size_t)(lane & 15) * ld + ((lane >> 4) * 8);
  union { float4 f[2]; v16h v; } u;
  u.f[0] = *reinterpret_cast<const float4*>(p);
  u.f[1] = *reinterpret_cast<const float4*>(p + 16);
  return u.v;
}
// B-matrix fragment (32x16 f16, KxN) gathered from K-contiguous B^T rows:
// lane n = lane&15, half h holds K = {h*8..h*8+7} U {16+h*8..16+h*8+7}
__device__ __forceinline__ v16h load_fragB(const _Float16* tile, int ld, int lane) {
  const _Float16* p = tile + (size_t)(lane & 15) * ld + ((lane >> 4) * 8);
  union { float4 f[2]; v16h v; } u;
  u.f[0] = *reinterpret_cast<const float4*>(p);
  u.f[1] = *reinterpret_cast<const float4*>(p + 16);
  return u.v;
}

__device__ __forceinline__ float gelu_exact(float x) {
  return 0.5f * x * (1.f + erff(x * 0.7071067811865475f));
}

// ---------------- weight transpose + f16 convert: [L,K,N] f32 -> [L,N,K] f16 ----------------
__global__ void k_cvt_transpose(const float* __restrict__ src, _Float16* __restrict__ dst,
                                int L, int K, int N) {
  int idx = blockIdx.x * blockDim.x + threadIdx.x;
  int tot = L * K * N;
  if (idx >= tot) return;
  int l = idx / (K * N);
  int rem = idx - l * K * N;
  int k = rem / N;
  int n = rem - k * N;
  dst[((size_t)l * N + n) * K + k] = (_Float16)src[idx];
}

// ---------------- input projection + LayerNorm + rotary ----------------
// src [B,M,T] f32, W [M,128], out x [B,T,128] f32. One 128-thread block per (t,b).
__global__ void k_inproj(const float* __restrict__ src, const float* __restrict__ W,
                         const float* __restrict__ bias,
                         const float* __restrict__ lng, const float* __restrict__ lnb,
                         const float* __restrict__ fcos, const float* __restrict__ fsin,
                         const float* __restrict__ scale,   // nullptr => pos = t
                         float* __restrict__ xout, int Msz, int T) {
  __shared__ float scol[512];
  __shared__ float snorm[128];
  __shared__ float sred[32];
  int t = blockIdx.x, b = blockIdx.y, c = threadIdx.x;
  for (int m = c; m < Msz; m += 128) scol[m] = src[((size_t)b * Msz + m) * T + t];
  __syncthreads();
  float acc = bias[c];
  for (int m = 0; m < Msz; ++m) acc = fmaf(scol[m], W[m * CC + c], acc);
  float mean = blockReduceSum(acc, sred) * (1.f / CC);
  float d = acc - mean;
  float var = blockReduceSum(d * d, sred) * (1.f / CC);
  snorm[c] = d * rsqrtf(var + 1e-5f) * lng[c] + lnb[c];
  __syncthreads();
  if (c < 64) {
    int pos = t;
    if (scale) {
      pos = (int)((float)t * scale[b]);
      if (pos > MAXPOS - 1) pos = MAXPOS - 1;
    }
    float cp = fcos[(size_t)pos * 64 + c] * POSA;
    float sp = fsin[(size_t)pos * 64 + c] * POSA;
    float x0 = snorm[2 * c], x1 = snorm[2 * c + 1];
    size_t o = ((size_t)b * T + t) * CC;
    xout[o + 2 * c]     = x0 * cp - x1 * sp;
    xout[o + 2 * c + 1] = x0 * sp + x1 * cp;
  }
}

// ---------------- depthwise conv7 (along T) + LayerNorm(1e-6) -> f16 ----------------
__global__ void k_dwln(const float* __restrict__ x, const float* __restrict__ dww,
                       const float* __restrict__ dwb,
                       const float* __restrict__ lng, const float* __restrict__ lnb,
                       _Float16* __restrict__ out, int T) {
  __shared__ float sred[32];
  int t = blockIdx.x, b = blockIdx.y, c = threadIdx.x;
  float acc = dwb[c];
#pragma unroll
  for (int j = 0; j < 7; ++j) {
    int tt = t + j - 3;
    if (tt >= 0 && tt < T) acc = fmaf(x[((size_t)b * T + tt) * CC + c], dww[c * 7 + j], acc);
  }
  float mean = blockReduceSum(acc, sred) * (1.f / CC);
  float d = acc - mean;
  float var = blockReduceSum(d * d, sred) * (1.f / CC);
  out[((size_t)b * T + t) * CC + c] = (_Float16)(d * rsqrtf(var + 1e-6f) * lng[c] + lnb[c]);
}

// ---------------- WMMA GEMM: pw1 (K=128 -> HID) + bias + GELU -> f16 ----------------
// Weight slice [128 rows x 128 K] staged into LDS via async global->LDS copies.
__global__ __launch_bounds__(256) void k_pw1(const _Float16* __restrict__ A,   // [B,T,128]
                                             const _Float16* __restrict__ W1T, // [384,128]
                                             const float* __restrict__ b1,
                                             _Float16* __restrict__ Hout, int T) {
  __shared__ __align__(16) _Float16 sW[128 * CC];   // 32 KB
  int lane = threadIdx.x & 31, w = threadIdx.x >> 5;
  int wm = w & 3, wn = w >> 2;
  int b = blockIdx.z;
  int m0 = blockIdx.x * 64 + wm * 16;
  int nblk = blockIdx.y * 128;        // this block's 128 weight rows
  // ---- stage weights: 32 KB = 256 threads x 8 x 16B, ASYNCcnt-tracked ----
  {
    unsigned lds0 = (unsigned)(unsigned long long)(uintptr_t)(void*)sW;
    const char* gw = (const char*)(W1T + (size_t)nblk * CC);
#pragma unroll
    for (int i = 0; i < 8; ++i) {
      unsigned off = ((unsigned)threadIdx.x + (unsigned)i * 256u) * 16u;
      unsigned ldsaddr = lds0 + off;
      unsigned long long ga = (unsigned long long)(uintptr_t)(gw + off);
      ASYNC_COPY_B128(ldsaddr, ga);
    }
    WAIT_ASYNCCNT0();
  }
  __syncthreads();

  const _Float16* Arow = A + ((size_t)b * T + m0) * CC;
  v8f acc[4] = {};
#pragma unroll
  for (int k0 = 0; k0 < CC; k0 += 32) {
    __builtin_prefetch(Arow + k0 + 64, 0, 1);
    v16h a = load_fragA(Arow + k0, CC, lane);
#pragma unroll
    for (int nt = 0; nt < 4; ++nt) {
      v16h bf = load_fragB(sW + (size_t)(wn * 64 + nt * 16) * CC + k0, CC, lane);
      acc[nt] = __builtin_amdgcn_wmma_f32_16x16x32_f16(false, a, false, bf, (short)0,
                                                       acc[nt], false, false);
    }
  }
  int rbase = (lane >> 4) * 8, nc = lane & 15;
#pragma unroll
  for (int nt = 0; nt < 4; ++nt) {
    int n = nblk + wn * 64 + nt * 16 + nc;
    float bias = b1[n];
#pragma unroll
    for (int i = 0; i < 8; ++i) {
      int m = m0 + rbase + i;
      float v = acc[nt][i] + bias;
      Hout[((size_t)b * T + m) * HIDC + n] = (_Float16)gelu_exact(v);
    }
  }
}

// ---------------- GRN stage 1: gx2[b,h] = sum_t H^2 ----------------
__global__ void k_gx2(const _Float16* __restrict__ H, float* __restrict__ gx2, int T) {
  int idx = blockIdx.x * blockDim.x + threadIdx.x;
  if (idx >= BB * HIDC) return;
  int b = idx / HIDC, h = idx - b * HIDC;
  const _Float16* p = H + (size_t)b * T * HIDC + h;
  float s = 0.f;
  for (int t = 0; t < T; ++t) {
    float v = (float)p[(size_t)t * HIDC];
    s = fmaf(v, v, s);
  }
  gx2[idx] = s;
}

// ---------------- GRN stage 2: nx = gx / (mean_h gx + 1e-6) ----------------
__global__ void k_nx(const float* __restrict__ gx2, float* __restrict__ nx) {
  __shared__ float sred[32];
  int b = blockIdx.x, h = threadIdx.x;
  float g = sqrtf(gx2[b * HIDC + h]);
  float mean = blockReduceSum(g, sred) * (1.f / HIDC);
  nx[b * HIDC + h] = g / (mean + 1e-6f);
}

// ---------------- GRN stage 3: H = g*(H*nx) + b + H (in place) ----------------
__global__ void k_grn(_Float16* __restrict__ H, const float* __restrict__ nx,
                      const float* __restrict__ gg, const float* __restrict__ gb, int T) {
  size_t tot = (size_t)BB * T * HIDC;
  for (size_t i = (size_t)blockIdx.x * blockDim.x + threadIdx.x; i < tot;
       i += (size_t)gridDim.x * blockDim.x) {
    int h = (int)(i % HIDC);
    int b = (int)(i / ((size_t)T * HIDC));
    float v = (float)H[i];
    H[i] = (_Float16)(gg[h] * (v * nx[b * HIDC + h]) + gb[h] + v);
  }
}

// ---------------- WMMA GEMM: pw2 (K=384 -> 128) + bias + residual into x f32 ----------------
__global__ __launch_bounds__(256) void k_pw2(const _Float16* __restrict__ H,   // [B,T,384]
                                             const _Float16* __restrict__ W2T, // [128,384]
                                             const float* __restrict__ b2,
                                             float* __restrict__ x, int T) {
  int lane = threadIdx.x & 31, w = threadIdx.x >> 5;
  int wm = w & 3, wn = w >> 2;
  int b = blockIdx.z;
  int m0 = blockIdx.x * 64 + wm * 16;
  int n0 = wn * 64;
  const _Float16* Arow = H + ((size_t)b * T + m0) * HIDC;
  v8f acc[4] = {};
  for (int k0 = 0; k0 < HIDC; k0 += 32) {
    __builtin_prefetch(Arow + k0 + 64, 0, 1);
    v16h a = load_fragA(Arow + k0, HIDC, lane);
#pragma unroll
    for (int nt = 0; nt < 4; ++nt) {
      v16h bf = load_fragB(W2T + (size_t)(n0 + nt * 16) * HIDC + k0, HIDC, lane);
      acc[nt] = __builtin_amdgcn_wmma_f32_16x16x32_f16(false, a, false, bf, (short)0,
                                                       acc[nt], false, false);
    }
  }
  int rbase = (lane >> 4) * 8, nc = lane & 15;
#pragma unroll
  for (int nt = 0; nt < 4; ++nt) {
    int n = n0 + nt * 16 + nc;
    float bias = b2[n];
#pragma unroll
    for (int i = 0; i < 8; ++i) {
      int m = m0 + rbase + i;
      size_t idx = ((size_t)b * T + m) * CC + n;
      x[idx] = x[idx] + acc[nt][i] + bias;
    }
  }
}

// ---------------- row squared-norm + f16 convert ----------------
__global__ void k_rownorm(const float* __restrict__ x, _Float16* __restrict__ x16,
                          float* __restrict__ rn, int T) {
  __shared__ float sred[32];
  int t = blockIdx.x, b = blockIdx.y, c = threadIdx.x;
  size_t o = ((size_t)b * T + t) * CC + c;
  float v = x[o];
  x16[o] = (_Float16)v;
  float s = blockReduceSum(v * v, sred);
  if (c == 0) rn[b * T + t] = s;
}

// ---------------- WMMA attention: scores + softmax ----------------
// q tile (16 x 128 f16 = 4 KB) staged once into LDS by wave 0 via async copies,
// shared by all 8 waves (replaces 64 redundant global fragment loads).
__global__ __launch_bounds__(256) void k_attn(const _Float16* __restrict__ q16,
                                              const _Float16* __restrict__ k16,
                                              const float* __restrict__ q2,
                                              const float* __restrict__ k2,
                                              float* __restrict__ out) {
  __shared__ __align__(16) _Float16 sQ[16 * CC];   // 4 KB
  __shared__ float ssc[16 * T2C];                  // 32 KB
  int lane = threadIdx.x & 31, w = threadIdx.x >> 5;
  int b = blockIdx.y;
  int m0 = blockIdx.x * 16;
  int n0 = w * 64;                   // 8 waves cover all 512 columns
  // ---- wave 0 stages the q tile: 4 KB = 32 lanes x 8 x 16B ----
  if (threadIdx.x < 32) {
    unsigned lds0 = (unsigned)(unsigned long long)(uintptr_t)(void*)sQ;
    const char* gq = (const char*)(q16 + ((size_t)b * T1C + m0) * CC);
#pragma unroll
    for (int i = 0; i < 8; ++i) {
      unsigned off = ((unsigned)threadIdx.x + (unsigned)i * 32u) * 16u;
      unsigned ldsaddr = lds0 + off;
      unsigned long long ga = (unsigned long long)(uintptr_t)(gq + off);
      ASYNC_COPY_B128(ldsaddr, ga);
    }
    WAIT_ASYNCCNT0();
  }
  __syncthreads();

  const _Float16* Brow = k16 + ((size_t)b * T2C + n0) * CC;
  v8f acc[4] = {};
#pragma unroll
  for (int k0 = 0; k0 < CC; k0 += 32) {
    __builtin_prefetch(Brow + k0 + 64, 0, 1);
    v16h a = load_fragA(sQ + k0, CC, lane);
#pragma unroll
    for (int nt = 0; nt < 4; ++nt) {
      v16h bf = load_fragB(Brow + (size_t)nt * 16 * CC + k0, CC, lane);
      acc[nt] = __builtin_amdgcn_wmma_f32_16x16x32_f16(false, a, false, bf, (short)0,
                                                       acc[nt], false, false);
    }
  }
  int rbase = (lane >> 4) * 8, nc = lane & 15;
#pragma unroll
  for (int nt = 0; nt < 4; ++nt) {
    int j = n0 + nt * 16 + nc;
    float k2v = k2[b * T2C + j];
#pragma unroll
    for (int i = 0; i < 8; ++i) {
      int ml = rbase + i;
      float q2v = q2[b * T1C + m0 + ml];
      ssc[ml * T2C + j] = -(q2v + k2v - 2.f * acc[nt][i]) * (1.f / CC);
    }
  }
  __syncthreads();
  // softmax: each 16-lane half-wave owns one row (8 waves * 2 rows = 16 rows)
  int r = w * 2 + (lane >> 4);
  int j0 = lane & 15;
  float mx = -1e30f;
  for (int j = j0; j < T2C; j += 16) mx = fmaxf(mx, ssc[r * T2C + j]);
#pragma unroll
  for (int m = 8; m; m >>= 1) mx = fmaxf(mx, __shfl_xor(mx, m, 32));
  float sum = 0.f;
  for (int j = j0; j < T2C; j += 16) {
    float e = __expf(ssc[r * T2C + j] - mx);
    ssc[r * T2C + j] = e;
    sum += e;
  }
#pragma unroll
  for (int m = 8; m; m >>= 1) sum += __shfl_xor(sum, m, 32);
  float inv = 1.f / sum;
  size_t o = ((size_t)b * T1C + m0 + r) * T2C;
  for (int j = j0; j < T2C; j += 16) out[o + j] = ssc[r * T2C + j] * inv;
}

// ---------------- host launcher ----------------
extern "C" void kernel_launch(void* const* d_in, const int* in_sizes, int n_in,
                              void* d_out, int out_size, void* d_ws, size_t ws_size,
                              hipStream_t stream) {
  (void)in_sizes; (void)n_in; (void)out_size; (void)ws_size;
  const float* queries   = (const float*)d_in[0];
  const float* keys      = (const float*)d_in[1];
  const float* tes       = (const float*)d_in[2];
  const float* key_in_w  = (const float*)d_in[3];
  const float* key_in_b  = (const float*)d_in[4];
  const float* key_ln_g  = (const float*)d_in[5];
  const float* key_ln_b  = (const float*)d_in[6];
  const float* q_in_w    = (const float*)d_in[7];
  const float* q_in_b    = (const float*)d_in[8];
  const float* q_ln_g    = (const float*)d_in[9];
  const float* q_ln_b    = (const float*)d_in[10];
  const float* kb_dw_w   = (const float*)d_in[11];
  const float* kb_dw_b   = (const float*)d_in[12];
  const float* kb_ln_g   = (const float*)d_in[13];
  const float* kb_ln_b   = (const float*)d_in[14];
  const float* kb_pw1_w  = (const float*)d_in[15];
  const float* kb_pw1_b  = (const float*)d_in[16];
  const float* kb_grn_g  = (const float*)d_in[17];
  const float* kb_grn_b  = (const float*)d_in[18];
  const float* kb_pw2_w  = (const float*)d_in[19];
  const float* kb_pw2_b  = (const float*)d_in[20];
  const float* qb_dw_w   = (const float*)d_in[21];
  const float* qb_dw_b   = (const float*)d_in[22];
  const float* qb_ln_g   = (const float*)d_in[23];
  const float* qb_ln_b   = (const float*)d_in[24];
  const float* qb_pw1_w  = (const float*)d_in[25];
  const float* qb_pw1_b  = (const float*)d_in[26];
  const float* qb_grn_g  = (const float*)d_in[27];
  const float* qb_grn_b  = (const float*)d_in[28];
  const float* qb_pw2_w  = (const float*)d_in[29];
  const float* qb_pw2_b  = (const float*)d_in[30];
  const float* fcos      = (const float*)d_in[31];
  const float* fsin      = (const float*)d_in[32];
  float* out = (float*)d_out;

  // ---- workspace carve (256B aligned) ----
  char* ws = (char*)d_ws;
  size_t off = 0;
  auto carve = [&](size_t bytes) -> char* {
    char* p = ws + off;
    off += (bytes + 255) & ~(size_t)255;
    return p;
  };
  float*    xq   = (float*)   carve((size_t)BB * T1C * CC * 4);
  float*    xk   = (float*)   carve((size_t)BB * T2C * CC * 4);
  _Float16* ln16 = (_Float16*)carve((size_t)BB * T1C * CC * 2);
  _Float16* Hbuf = (_Float16*)carve((size_t)BB * T1C * HIDC * 2);
  _Float16* W1Tk = (_Float16*)carve((size_t)3 * HIDC * CC * 2);
  _Float16* W2Tk = (_Float16*)carve((size_t)3 * CC * HIDC * 2);
  _Float16* W1Tq = (_Float16*)carve((size_t)3 * HIDC * CC * 2);
  _Float16* W2Tq = (_Float16*)carve((size_t)3 * CC * HIDC * 2);
  float*    gx2  = (float*)   carve((size_t)BB * HIDC * 4);
  float*    nxv  = (float*)   carve((size_t)BB * HIDC * 4);
  _Float16* q16  = (_Float16*)carve((size_t)BB * T1C * CC * 2);
  _Float16* k16  = (_Float16*)carve((size_t)BB * T2C * CC * 2);
  float*    q2b  = (float*)   carve((size_t)BB * T1C * 4);
  float*    k2b  = (float*)   carve((size_t)BB * T2C * 4);

  // ---- weight transpose/convert ----
  {
    int tot = 3 * CC * HIDC;
    k_cvt_transpose<<<(tot + 255) / 256, 256, 0, stream>>>(kb_pw1_w, W1Tk, 3, CC, HIDC);
    k_cvt_transpose<<<(tot + 255) / 256, 256, 0, stream>>>(kb_pw2_w, W2Tk, 3, HIDC, CC);
    k_cvt_transpose<<<(tot + 255) / 256, 256, 0, stream>>>(qb_pw1_w, W1Tq, 3, CC, HIDC);
    k_cvt_transpose<<<(tot + 255) / 256, 256, 0, stream>>>(qb_pw2_w, W2Tq, 3, HIDC, CC);
  }

  // ---- input projections + LN + rotary ----
  k_inproj<<<dim3(T2C, BB), 128, 0, stream>>>(keys, key_in_w, key_in_b, key_ln_g, key_ln_b,
                                              fcos, fsin, tes, xk, NTEXT, T2C);
  k_inproj<<<dim3(T1C, BB), 128, 0, stream>>>(queries, q_in_w, q_in_b, q_ln_g, q_ln_b,
                                              fcos, fsin, nullptr, xq, NMEL, T1C);

  // ---- ConvNeXtV2 stacks ----
  for (int l = 0; l < 3; ++l) {  // key stream, T = 512
    k_dwln<<<dim3(T2C, BB), 128, 0, stream>>>(xk, kb_dw_w + l * CC * 7, kb_dw_b + l * CC,
                                              kb_ln_g + l * CC, kb_ln_b + l * CC, ln16, T2C);
    k_pw1<<<dim3(T2C / 64, HIDC / 128, BB), 256, 0, stream>>>(
        ln16, W1Tk + (size_t)l * HIDC * CC, kb_pw1_b + l * HIDC, Hbuf, T2C);
    k_gx2<<<(BB * HIDC + 255) / 256, 256, 0, stream>>>(Hbuf, gx2, T2C);
    k_nx<<<BB, HIDC, 0, stream>>>(gx2, nxv);
    k_grn<<<1024, 256, 0, stream>>>(Hbuf, nxv, kb_grn_g + l * HIDC, kb_grn_b + l * HIDC, T2C);
    k_pw2<<<dim3(T2C / 64, 1, BB), 256, 0, stream>>>(
        Hbuf, W2Tk + (size_t)l * CC * HIDC, kb_pw2_b + l * CC, xk, T2C);
  }
  for (int l = 0; l < 3; ++l) {  // query stream, T = 2048
    k_dwln<<<dim3(T1C, BB), 128, 0, stream>>>(xq, qb_dw_w + l * CC * 7, qb_dw_b + l * CC,
                                              qb_ln_g + l * CC, qb_ln_b + l * CC, ln16, T1C);
    k_pw1<<<dim3(T1C / 64, HIDC / 128, BB), 256, 0, stream>>>(
        ln16, W1Tq + (size_t)l * HIDC * CC, qb_pw1_b + l * HIDC, Hbuf, T1C);
    k_gx2<<<(BB * HIDC + 255) / 256, 256, 0, stream>>>(Hbuf, gx2, T1C);
    k_nx<<<BB, HIDC, 0, stream>>>(gx2, nxv);
    k_grn<<<1024, 256, 0, stream>>>(Hbuf, nxv, qb_grn_g + l * HIDC, qb_grn_b + l * HIDC, T1C);
    k_pw2<<<dim3(T1C / 64, 1, BB), 256, 0, stream>>>(
        Hbuf, W2Tq + (size_t)l * CC * HIDC, qb_pw2_b + l * CC, xq, T1C);
  }

  // ---- row norms + f16 convert ----
  k_rownorm<<<dim3(T2C, BB), 128, 0, stream>>>(xk, k16, k2b, T2C);
  k_rownorm<<<dim3(T1C, BB), 128, 0, stream>>>(xq, q16, q2b, T1C);

  // ---- attention ----
  k_attn<<<dim3(T1C / 16, BB), 256, 0, stream>>>(q16, k16, q2b, k2b, out);
}
